// AttentionCharRNN_38525856645856
// MI455X (gfx1250) — matrix operand order repaired and checked by reading, not compiled, not run on hardware
//
#include <hip/hip_runtime.h>
#include <hip/hip_bf16.h>

// Bidirectional LSTM + additive attention for MI455X (gfx1250).
// Matmuls: v_wmma_f32_16x16x32_bf16 (wave32, f32 accumulate).
// CDNA5 data movement: TDM tensor_load_to_lds stages shared WMMA operands
// (W_hh slice + h matrix per step; attention A-tiles) into LDS, synced with
// s_wait_tensorcnt + workgroup barrier; WMMA operands then come from ds loads.

#define TDIM 512
#define BDIM 64
#define EDIM 256
#define HDIM 512
#define G4   2048   // 4*H
#define H2   1024   // 2*H
#define TB   (TDIM*BDIM)

typedef __attribute__((ext_vector_type(16))) __bf16 v16bf;
typedef __attribute__((ext_vector_type(8)))  __bf16 bf8;
typedef __attribute__((ext_vector_type(8)))  float  v8f;
typedef __attribute__((ext_vector_type(4)))  unsigned int u32x4;
typedef __attribute__((ext_vector_type(8)))  int i32x8;
typedef __attribute__((ext_vector_type(4)))  int i32x4;

#define HAVE_TDM __has_builtin(__builtin_amdgcn_tensor_load_to_lds)

// ---- TDM 2D bf16 tile load: rows x width elements, row stride in elements ----
// D# per CDNA5 ISA 8.3/8.4: group0 = {count=1 | lds_addr | global_addr(57b) | type=2},
// group1 = {mask=0,data_size=1(2B) | tensor_dim0/1 | tile_dim0/1 | dim0_stride}, groups 2/3 zero (2D).
// This toolchain's builtin is the 6-arg (clang-23 / therock-10.0) form.
#if HAVE_TDM
__device__ __forceinline__ void tdm_load_bf16_2d(unsigned lds_byte_off, const __bf16* gptr,
                                                 unsigned width, unsigned rows, unsigned stride) {
  unsigned long long ga = (unsigned long long)gptr;
  u32x4 g0;
  g0[0] = 1u;                                                 // count=1 (valid user D#)
  g0[1] = lds_byte_off;                                       // lds_addr (bytes)
  g0[2] = (unsigned)ga;                                       // global_addr[31:0]
  g0[3] = (unsigned)((ga >> 32) & 0x1FFFFFFu) | (2u << 30);   // global_addr[56:32] | type=2
  i32x8 g1;
  g1[0] = (int)(1u << 16);                                    // wg_mask=0, data_size=1 (2 bytes)
  g1[1] = (int)((width & 0xFFFFu) << 16);                     // tensor_dim0[15:0]
  g1[2] = (int)((width >> 16) | ((rows & 0xFFFFu) << 16));    // tensor_dim0[31:16] | tensor_dim1[15:0]
  g1[3] = (int)((rows >> 16) | ((width & 0xFFFFu) << 16));    // tensor_dim1[31:16] | tile_dim0
  g1[4] = (int)(rows & 0xFFFFu);                              // tile_dim1 | tile_dim2=0
  g1[5] = (int)stride;                                        // tensor_dim0_stride[31:0]
  g1[6] = 0;
  g1[7] = 0;
  i32x4 z4 = {0, 0, 0, 0};
  i32x8 z8 = {0, 0, 0, 0, 0, 0, 0, 0};
  __builtin_amdgcn_tensor_load_to_lds(g0, g1, z4, z4, z8, 0);
}
#endif

// ---- WMMA tile loaders (ISA 7.12.2 layouts, wave32) ----
// A (16x32, 16-bit): lanes 0-15 M=lane, K={0..7,16..23}; lanes 16-31 M=lane-16, K={8..15,24..31}
__device__ __forceinline__ v16bf load_tileA(const __bf16* __restrict__ src, int row0, int k0, int ld) {
  int lane = threadIdx.x & 31;
  const __bf16* p = src + (size_t)(row0 + (lane & 15)) * ld + k0 + ((lane >> 4) << 3);
  bf8 lo = *(const bf8*)(p);
  bf8 hi = *(const bf8*)(p + 16);
  v16bf a;
#pragma unroll
  for (int i = 0; i < 8; ++i) { a[i] = lo[i]; a[i + 8] = hi[i]; }
  return a;
}
// B (32x16, 16-bit): lane = N, lanes 0-15 hold K={0..7,16..23}, lanes 16-31 hold K={8..15,24..31}.
// src is "N-major": B[k][n] = src[(n0+n)*ld + k0 + k]
__device__ __forceinline__ v16bf load_tileB(const __bf16* __restrict__ src, int n0, int k0, int ld) {
  int lane = threadIdx.x & 31;
  const __bf16* p = src + (size_t)(n0 + (lane & 15)) * ld + k0 + ((lane >> 4) << 3);
  bf8 lo = *(const bf8*)(p);
  bf8 hi = *(const bf8*)(p + 16);
  v16bf b;
#pragma unroll
  for (int i = 0; i < 8; ++i) { b[i] = lo[i]; b[i + 8] = hi[i]; }
  return b;
}

__device__ __forceinline__ float sigmoidf_(float x) { return 1.0f / (1.0f + __expf(-x)); }

// ---- small utility kernels ----
__global__ void k_cast_bf16(const float* __restrict__ s, __bf16* __restrict__ d, int n) {
  int i = blockIdx.x * blockDim.x + threadIdx.x;
  if (i < n) d[i] = (__bf16)s[i];
}
__global__ void k_transpose_bf16(const float* __restrict__ W, __bf16* __restrict__ Wt, int rows, int cols) {
  int i = blockIdx.x * blockDim.x + threadIdx.x;
  if (i < rows * cols) {
    int c = i / rows, r = i % rows;
    Wt[i] = (__bf16)W[(size_t)r * cols + c];
  }
}
__global__ void k_init_dir(const float* __restrict__ h0, const float* __restrict__ c0, int dir,
                           __bf16* __restrict__ hbf, float* __restrict__ c) {
  int i = blockIdx.x * blockDim.x + threadIdx.x;
  if (i < BDIM * HDIM) {
    hbf[i] = (__bf16)h0[dir * BDIM * HDIM + i];
    c[i]   = c0[dir * BDIM * HDIM + i];
  }
}
__global__ void k_zero(float* p, int n) {
  int i = blockIdx.x * blockDim.x + threadIdx.x;
  if (i < n) p[i] = 0.f;
}

// ---- input projection GEMM: xp[t*B+b, :] = embed[tokens] @ W_ih^T + bias ----
__global__ __launch_bounds__(128) void k_proj(const int* __restrict__ tokens,
        const __bf16* __restrict__ tablebf, const __bf16* __restrict__ wihbf,
        const float* __restrict__ bias, float* __restrict__ xp, int reverse) {
  int wave = threadIdx.x >> 5, lane = threadIdx.x & 31;
  int r0 = blockIdx.x * 64 + wave * 16;
  int nblk = blockIdx.y * 64;
  int r = r0 + (lane & 15);
  int tok_idx = reverse ? ((((TDIM - 1) - (r >> 6)) << 6) | (r & 63)) : r;
  const __bf16* arow = tablebf + (size_t)tokens[tok_idx] * EDIM + ((lane >> 4) << 3);

  v8f acc[4];
#pragma unroll
  for (int j = 0; j < 4; ++j) {
    float bv = bias[nblk + j * 16 + (lane & 15)];
#pragma unroll
    for (int q = 0; q < 8; ++q) acc[j][q] = bv;
  }
#pragma unroll
  for (int k0 = 0; k0 < EDIM; k0 += 32) {
    bf8 lo = *(const bf8*)(arow + k0);
    bf8 hi = *(const bf8*)(arow + k0 + 16);
    v16bf a;
#pragma unroll
    for (int i = 0; i < 8; ++i) { a[i] = lo[i]; a[i + 8] = hi[i]; }
#pragma unroll
    for (int j = 0; j < 4; ++j) {
      v16bf b = load_tileB(wihbf, nblk + j * 16, k0, EDIM);
      acc[j] = __builtin_amdgcn_wmma_f32_16x16x32_bf16(false, a, false, b, (short)0, acc[j], false, false);
    }
  }
#pragma unroll
  for (int j = 0; j < 4; ++j) {
    int col = nblk + j * 16 + (lane & 15);
#pragma unroll
    for (int q = 0; q < 8; ++q) {
      int m = r0 + q + ((lane >> 4) << 3);
      xp[(size_t)m * G4 + col] = acc[j][q];
    }
  }
}

// ---- one LSTM timestep ----
// grid 32 blocks x 128 threads. Block nb owns 16 h-columns; wave w owns 16 batch rows and
// computes all 4 gate tiles -> cell update fully in-register.
// TDM stages this WG's W_hh slice (4 x 16 rows x 512) and the full h matrix into LDS.
__global__ __launch_bounds__(128) void k_lstm_step(const float* __restrict__ xp,
        const __bf16* __restrict__ whhbf, const __bf16* __restrict__ hin,
        __bf16* __restrict__ hout, float* __restrict__ c,
        float* __restrict__ outf, __bf16* __restrict__ outbf, int t, int reverse) {
  __shared__ __bf16 smem[64 * HDIM + BDIM * HDIM];  // W slice (64KB) + h (64KB)
  int wave = threadIdx.x >> 5, lane = threadIdx.x & 31;
  int m0 = wave * 16;
  int nb = blockIdx.x * 16;
  const float* xprow = xp + (size_t)t * BDIM * G4;

#if HAVE_TDM
  {
    unsigned lds_base = (unsigned)(unsigned long long)(&smem[0]);
    // each wave stages one gate's 16 W_hh rows; wave 0 also stages h
    tdm_load_bf16_2d(lds_base + (unsigned)(wave * 16 * HDIM * 2),
                     whhbf + (size_t)(wave * HDIM + nb) * HDIM, HDIM, 16, HDIM);
    if (wave == 0)
      tdm_load_bf16_2d(lds_base + (unsigned)(64 * HDIM * 2), hin, HDIM, BDIM, HDIM);
    __builtin_amdgcn_s_wait_tensorcnt(0);
  }
  __syncthreads();
#else
  for (int i = threadIdx.x; i < 64 * HDIM / 8; i += 128) {
    int lr = i / (HDIM / 8), cc = (i % (HDIM / 8)) * 8;
    int g = lr >> 4, n = lr & 15;
    ((bf8*)smem)[i] = *(const bf8*)(whhbf + (size_t)(g * HDIM + nb + n) * HDIM + cc);
  }
  for (int i = threadIdx.x; i < BDIM * HDIM / 8; i += 128)
    ((bf8*)(smem + 64 * HDIM))[i] = ((const bf8*)hin)[i];
  __syncthreads();
#endif

  v8f acc[4];
#pragma unroll
  for (int g = 0; g < 4; ++g) {
    int col = g * HDIM + nb + (lane & 15);
#pragma unroll
    for (int q = 0; q < 8; ++q) {
      int m = m0 + q + ((lane >> 4) << 3);
      acc[g][q] = xprow[(size_t)m * G4 + col];
    }
  }
#pragma unroll 4
  for (int k0 = 0; k0 < HDIM; k0 += 32) {
    // A from LDS-staged h
    v16bf a;
    {
      const __bf16* p = &smem[64 * HDIM + (m0 + (lane & 15)) * HDIM + k0 + ((lane >> 4) << 3)];
      bf8 lo = *(const bf8*)(p);
      bf8 hi = *(const bf8*)(p + 16);
#pragma unroll
      for (int i = 0; i < 8; ++i) { a[i] = lo[i]; a[i + 8] = hi[i]; }
    }
#pragma unroll
    for (int g = 0; g < 4; ++g) {
      // B from LDS-staged W slice: local row = g*16 + n
      v16bf b;
      const __bf16* p = &smem[(g * 16 + (lane & 15)) * HDIM + k0 + ((lane >> 4) << 3)];
      bf8 lo = *(const bf8*)(p);
      bf8 hi = *(const bf8*)(p + 16);
#pragma unroll
      for (int i = 0; i < 8; ++i) { b[i] = lo[i]; b[i + 8] = hi[i]; }
      acc[g] = __builtin_amdgcn_wmma_f32_16x16x32_bf16(false, a, false, b, (short)0, acc[g], false, false);
    }
  }

  int t_out = reverse ? (TDIM - 1 - t) : t;
  int dcol = reverse ? HDIM : 0;
  int col = nb + (lane & 15);
#pragma unroll
  for (int q = 0; q < 8; ++q) {
    int m = m0 + q + ((lane >> 4) << 3);
    float iv = sigmoidf_(acc[0][q]);
    float fv = sigmoidf_(acc[1][q]);
    float gv = tanhf(acc[2][q]);
    float ov = sigmoidf_(acc[3][q]);
    size_t cidx = (size_t)m * HDIM + col;
    float cn = fv * c[cidx] + iv * gv;
    c[cidx] = cn;
    float hv = ov * tanhf(cn);
    size_t oidx = (size_t)t_out * BDIM * H2 + (size_t)m * H2 + dcol + col;
    outf[oidx] = hv;
    __bf16 hb = (__bf16)hv;
    outbf[oidx] = hb;
    hout[(size_t)m * HDIM + col] = hb;
  }
}

// ---- attention scores: scores[r] = sum_n tanh((out@W_W)[r,n] + b_att[n]) * w_proj[n] ----
// TDM stages the shared A tile (16 rows x 1024 bf16) into LDS once per block.
__global__ __launch_bounds__(128) void k_scores(const __bf16* __restrict__ outbf,
        const __bf16* __restrict__ wwT, const float* __restrict__ b_att,
        const float* __restrict__ wproj, float* __restrict__ scores) {
  __shared__ __bf16 sA[16 * H2];  // 32KB
  int wave = threadIdx.x >> 5, lane = threadIdx.x & 31;
  int r0 = blockIdx.x * 16;
  int n0 = wave * 256;

#if HAVE_TDM
  if (wave == 0) {
    unsigned lds_base = (unsigned)(unsigned long long)(&sA[0]);
    tdm_load_bf16_2d(lds_base, outbf + (size_t)r0 * H2, H2, 16, H2);
    __builtin_amdgcn_s_wait_tensorcnt(0);
  }
  __syncthreads();
#else
  for (int i = threadIdx.x; i < 16 * H2 / 8; i += 128)
    ((bf8*)sA)[i] = *(const bf8*)(outbf + (size_t)r0 * H2 + i * 8);
  __syncthreads();
#endif

  v8f acc[16];
#pragma unroll
  for (int j = 0; j < 16; ++j) {
    float bv = b_att[n0 + j * 16 + (lane & 15)];
#pragma unroll
    for (int q = 0; q < 8; ++q) acc[j][q] = bv;
  }
  for (int k0 = 0; k0 < H2; k0 += 32) {
    v16bf a;
    {
      const __bf16* p = &sA[(lane & 15) * H2 + k0 + ((lane >> 4) << 3)];
      bf8 lo = *(const bf8*)(p);
      bf8 hi = *(const bf8*)(p + 16);
#pragma unroll
      for (int i = 0; i < 8; ++i) { a[i] = lo[i]; a[i + 8] = hi[i]; }
    }
#pragma unroll
    for (int j = 0; j < 16; ++j) {
      v16bf b = load_tileB(wwT, n0 + j * 16, k0, H2);
      acc[j] = __builtin_amdgcn_wmma_f32_16x16x32_bf16(false, a, false, b, (short)0, acc[j], false, false);
    }
  }
  float p[8];
#pragma unroll
  for (int q = 0; q < 8; ++q) p[q] = 0.f;
#pragma unroll
  for (int j = 0; j < 16; ++j) {
    float wp = wproj[n0 + j * 16 + (lane & 15)];
#pragma unroll
    for (int q = 0; q < 8; ++q) p[q] += tanhf(acc[j][q]) * wp;
  }
#pragma unroll
  for (int q = 0; q < 8; ++q) {
#pragma unroll
    for (int msk = 1; msk < 16; msk <<= 1) p[q] += __shfl_xor(p[q], msk, 32);
  }
  if ((lane & 15) == 0) {
#pragma unroll
    for (int q = 0; q < 8; ++q) {
      int m = r0 + q + ((lane >> 4) << 3);
      atomicAdd(&scores[m], p[q]);
    }
  }
}

// ---- softmax over time (per batch column) ----
__global__ __launch_bounds__(512) void k_softmax(const float* __restrict__ scores, float* __restrict__ attn) {
  __shared__ float sm[TDIM];
  int b = blockIdx.x, t = threadIdx.x;
  float s = scores[t * BDIM + b];
  sm[t] = s; __syncthreads();
  for (int o = 256; o > 0; o >>= 1) { if (t < o) sm[t] = fmaxf(sm[t], sm[t + o]); __syncthreads(); }
  float mx = sm[0]; __syncthreads();
  float e = __expf(s - mx);
  sm[t] = e; __syncthreads();
  for (int o = 256; o > 0; o >>= 1) { if (t < o) sm[t] += sm[t + o]; __syncthreads(); }
  attn[t * BDIM + b] = e / sm[0];
}

// ---- char_vectors[b,k] = sum_t attn[t,b] * out[t,b,k] ----
__global__ __launch_bounds__(256) void k_wsum(const float* __restrict__ attn,
        const float* __restrict__ outf, float* __restrict__ out) {
  int b = blockIdx.x;
  int k = blockIdx.y * 256 + threadIdx.x;
  float acc = 0.f;
  for (int t = 0; t < TDIM; ++t)
    acc += attn[t * BDIM + b] * outf[(size_t)t * BDIM * H2 + (size_t)b * H2 + k];
  out[b * H2 + k] = acc;
}

extern "C" void kernel_launch(void* const* d_in, const int* in_sizes, int n_in,
                              void* d_out, int out_size, void* d_ws, size_t ws_size,
                              hipStream_t stream) {
  const int*   tokens = (const int*)d_in[0];
  const float* h0     = (const float*)d_in[1];
  const float* c0     = (const float*)d_in[2];
  const float* table  = (const float*)d_in[3];
  const float* Wih_f  = (const float*)d_in[4];
  const float* Whh_f  = (const float*)d_in[5];
  const float* b_f    = (const float*)d_in[6];
  const float* Wih_b  = (const float*)d_in[7];
  const float* Whh_b  = (const float*)d_in[8];
  const float* b_b    = (const float*)d_in[9];
  const float* W_W    = (const float*)d_in[10];
  const float* b_att  = (const float*)d_in[11];
  const float* wproj  = (const float*)d_in[12];
  float* out = (float*)d_out;

  char* ws = (char*)d_ws;
  size_t off = 0;
  auto take = [&](size_t bytes) -> char* {
    char* p = ws + off;
    off = (off + bytes + 255) & ~(size_t)255;
    return p;
  };
  float*  xp     = (float*) take((size_t)TB * G4 * 4);
  float*  outf   = (float*) take((size_t)TB * H2 * 4);
  __bf16* outbf  = (__bf16*)take((size_t)TB * H2 * 2);
  __bf16* tbf    = (__bf16*)take((size_t)256 * EDIM * 2);
  __bf16* wihfb  = (__bf16*)take((size_t)G4 * EDIM * 2);
  __bf16* wihbb  = (__bf16*)take((size_t)G4 * EDIM * 2);
  __bf16* whhfb  = (__bf16*)take((size_t)G4 * HDIM * 2);
  __bf16* whhbb  = (__bf16*)take((size_t)G4 * HDIM * 2);
  __bf16* wwT    = (__bf16*)take((size_t)H2 * H2 * 2);
  __bf16* hA     = (__bf16*)take((size_t)BDIM * HDIM * 2);
  __bf16* hB     = (__bf16*)take((size_t)BDIM * HDIM * 2);
  float*  cbuf   = (float*) take((size_t)BDIM * HDIM * 4);
  float*  scores = (float*) take((size_t)TB * 4);
  float*  attn   = (float*) take((size_t)TB * 4);

  k_cast_bf16<<<(256 * EDIM + 255) / 256, 256, 0, stream>>>(table, tbf, 256 * EDIM);
  k_cast_bf16<<<(G4 * EDIM + 255) / 256, 256, 0, stream>>>(Wih_f, wihfb, G4 * EDIM);
  k_cast_bf16<<<(G4 * EDIM + 255) / 256, 256, 0, stream>>>(Wih_b, wihbb, G4 * EDIM);
  k_cast_bf16<<<(G4 * HDIM + 255) / 256, 256, 0, stream>>>(Whh_f, whhfb, G4 * HDIM);
  k_cast_bf16<<<(G4 * HDIM + 255) / 256, 256, 0, stream>>>(Whh_b, whhbb, G4 * HDIM);
  k_transpose_bf16<<<(H2 * H2 + 255) / 256, 256, 0, stream>>>(W_W, wwT, H2, H2);
  k_zero<<<(TB + 255) / 256, 256, 0, stream>>>(scores, TB);

  dim3 pg(TB / 64, G4 / 64);

  // forward direction
  k_proj<<<pg, 128, 0, stream>>>(tokens, tbf, wihfb, b_f, xp, 0);
  k_init_dir<<<(BDIM * HDIM + 255) / 256, 256, 0, stream>>>(h0, c0, 0, hA, cbuf);
  for (int t = 0; t < TDIM; ++t) {
    __bf16* hin  = (t & 1) ? hB : hA;
    __bf16* hout = (t & 1) ? hA : hB;
    k_lstm_step<<<HDIM / 16, 128, 0, stream>>>(xp, whhfb, hin, hout, cbuf, outf, outbf, t, 0);
  }
  // backward direction (xp buffer reused)
  k_proj<<<pg, 128, 0, stream>>>(tokens, tbf, wihbb, b_b, xp, 1);
  k_init_dir<<<(BDIM * HDIM + 255) / 256, 256, 0, stream>>>(h0, c0, 1, hA, cbuf);
  for (int t = 0; t < TDIM; ++t) {
    __bf16* hin  = (t & 1) ? hB : hA;
    __bf16* hout = (t & 1) ? hA : hB;
    k_lstm_step<<<HDIM / 16, 128, 0, stream>>>(xp, whhbb, hin, hout, cbuf, outf, outbf, t, 1);
  }

  // attention
  k_scores<<<TB / 16, 128, 0, stream>>>(outbf, wwT, b_att, wproj, scores);
  k_softmax<<<BDIM, TDIM, 0, stream>>>(scores, attn);
  k_wsum<<<dim3(BDIM, H2 / 256), 256, 0, stream>>>(attn, outf, out);
}
